// SecBertSelfAttention_63385127354823
// MI455X (gfx1250) — matrix operand order repaired and checked by reading, not compiled, not run on hardware
//
#include <hip/hip_runtime.h>
#include <hip/hip_bf16.h>
#include <stdint.h>

// ---------------------------------------------------------------------------
// SecBertSelfAttention for MI455X (gfx1250), wave32 + v_wmma_f32_16x16x32_f16.
// Pipeline: fp32->f16 repack  ->  fused QKV GEMM (WMMA, software-pipelined)
// -> flash attention (WMMA QK^T, online softmax, WMMA PV), fp32 output.
// ---------------------------------------------------------------------------

typedef _Float16 v16h __attribute__((ext_vector_type(16)));
typedef _Float16 v8h  __attribute__((ext_vector_type(8)));
typedef float    v8f  __attribute__((ext_vector_type(8)));

static __device__ __forceinline__ v16h cat16(v8h lo, v8h hi) {
    return __builtin_shufflevector(lo, hi, 0,1,2,3,4,5,6,7,8,9,10,11,12,13,14,15);
}
static __device__ __forceinline__ v8h ld8h(const _Float16* p) { return *(const v8h*)p; }
static __device__ __forceinline__ v8f zero8() {
    v8f z;
#pragma unroll
    for (int i = 0; i < 8; ++i) z[i] = 0.0f;
    return z;
}
#define WMMA_F16(a, b, c) \
    __builtin_amdgcn_wmma_f32_16x16x32_f16(false, (a), false, (b), (short)0, (c), false, false)

// Problem constants
#define BATCH    4
#define SEQ      1024
#define HID      1024
#define NHEAD    16
#define HDIM     64
#define MROWS    (BATCH * SEQ)          // 4096
#define NCOLS    (3 * HID)              // 3072 (Q|K|V fused)

// ---------------------------------------------------------------------------
// Kernel 1: fp32 -> f16 repack of activations and the three weight matrices.
// W kept in [N][K] row-major == column-major B operand for y = x @ W^T.
// ---------------------------------------------------------------------------
__global__ void cvt_kernel(const float* __restrict__ X,
                           const float* __restrict__ Wq,
                           const float* __restrict__ Wk,
                           const float* __restrict__ Wv,
                           _Float16* __restrict__ Xh,
                           _Float16* __restrict__ Wh) {
    const int NX = MROWS * HID;          // 4M
    const int NW = HID * HID;            // 1M (power of two: 2^20)
    const int total = NX + 3 * NW;
    int stride = gridDim.x * blockDim.x;
    for (int i = blockIdx.x * blockDim.x + threadIdx.x; i < total; i += stride) {
        if (i < NX) {
            __builtin_prefetch(&X[i] + stride, 0, 1);   // global_prefetch_b8
            Xh[i] = (_Float16)X[i];
        } else {
            int j = i - NX;
            int w = j >> 20;
            int o = j & (NW - 1);
            const float* src = (w == 0) ? Wq : (w == 1) ? Wk : Wv;
            Wh[j] = (_Float16)src[o];
        }
    }
}

// ---------------------------------------------------------------------------
// Kernel 2: fused QKV GEMM. One wave computes a 16x64 tile of [4096 x 3072].
// Software-pipelined K-loop (32 steps of v_wmma_f32_16x16x32_f16): fragments
// for step k+1 are loaded into fresh registers while the 4 WMMAs of step k
// execute, so the backend can use partial s_wait_loadcnt instead of full
// serialization. Epilogue: +bias, Q pre-scaled by 1/sqrt(64), scatter into
// head-major Q/K [bh][S][64] and transposed V [bh][64][S] (all f16).
// ---------------------------------------------------------------------------
__global__ void __launch_bounds__(256)
qkv_kernel(const _Float16* __restrict__ Xh, const _Float16* __restrict__ Wh,
           const float* __restrict__ bq, const float* __restrict__ bk,
           const float* __restrict__ bv,
           _Float16* __restrict__ Qh, _Float16* __restrict__ Kh,
           _Float16* __restrict__ Vt) {
    const int NT64 = NCOLS / 64;                      // 48 tile-columns
    int wid  = (blockIdx.x * blockDim.x + threadIdx.x) >> 5;
    int lane = threadIdx.x & 31;
    int mt   = wid / NT64;                            // 0..255
    int nt   = wid % NT64;                            // 0..47
    int half = lane >> 4;
    int r    = lane & 15;

    const _Float16* arow = Xh + (size_t)(mt * 16 + r) * HID;
    const _Float16* b0 = Wh + (size_t)(nt * 64 +  0 + r) * HID + half * 16;
    const _Float16* b1 = Wh + (size_t)(nt * 64 + 16 + r) * HID + half * 16;
    const _Float16* b2 = Wh + (size_t)(nt * 64 + 32 + r) * HID + half * 16;
    const _Float16* b3 = Wh + (size_t)(nt * 64 + 48 + r) * HID + half * 16;

    v8f acc0 = zero8(), acc1 = zero8(), acc2 = zero8(), acc3 = zero8();

    // --- pipeline prologue: stage k0 = 0 ---
    v16h aC  = cat16(ld8h(arow + half * 16), ld8h(arow + half * 16 + 8));
    v16h bC0 = cat16(ld8h(b0), ld8h(b0 + 8));
    v16h bC1 = cat16(ld8h(b1), ld8h(b1 + 8));
    v16h bC2 = cat16(ld8h(b2), ld8h(b2 + 8));
    v16h bC3 = cat16(ld8h(b3), ld8h(b3 + 8));

    for (int k0 = 32; k0 < HID; k0 += 32) {
        // issue next-stage loads first (independent registers)
        v16h aN  = cat16(ld8h(arow + k0 + half * 16),
                         ld8h(arow + k0 + half * 16 + 8));
        v16h bN0 = cat16(ld8h(b0 + k0), ld8h(b0 + k0 + 8));
        v16h bN1 = cat16(ld8h(b1 + k0), ld8h(b1 + k0 + 8));
        v16h bN2 = cat16(ld8h(b2 + k0), ld8h(b2 + k0 + 8));
        v16h bN3 = cat16(ld8h(b3 + k0), ld8h(b3 + k0 + 8));
        // consume current stage: 4 independent WMMAs, back-to-back
        acc0 = WMMA_F16(aC, bC0, acc0);
        acc1 = WMMA_F16(aC, bC1, acc1);
        acc2 = WMMA_F16(aC, bC2, acc2);
        acc3 = WMMA_F16(aC, bC3, acc3);
        aC = aN; bC0 = bN0; bC1 = bN1; bC2 = bN2; bC3 = bN3;
    }
    acc0 = WMMA_F16(aC, bC0, acc0);
    acc1 = WMMA_F16(aC, bC1, acc1);
    acc2 = WMMA_F16(aC, bC2, acc2);
    acc3 = WMMA_F16(aC, bC3, acc3);

    // Epilogue: C layout lane=N, vgpr v -> row v + 8*half
    int mbase = mt * 16;
    int b     = mbase >> 10;
    v8f accs[4] = {acc0, acc1, acc2, acc3};
#pragma unroll
    for (int t = 0; t < 4; ++t) {
        int ng  = nt * 64 + t * 16 + r;     // global N in 0..3071
        int reg = ng >> 10;                 // 0=Q 1=K 2=V (uniform per tile)
        int nw  = ng & 1023;
        int h   = nw >> 6, d = nw & 63;
        float bias  = (reg == 0) ? bq[nw] : (reg == 1) ? bk[nw] : bv[nw];
        float scale = (reg == 0) ? 0.125f : 1.0f;   // fold HEAD_DIM^-0.5 into Q
        int bh = b * NHEAD + h;
#pragma unroll
        for (int v = 0; v < 8; ++v) {
            int row = v + half * 8;
            int s   = (mbase + row) & 1023;
            float val = (accs[t][v] + bias) * scale;
            if (reg < 2) {
                _Float16* dst = (reg == 0) ? Qh : Kh;
                dst[((size_t)bh * SEQ + s) * HDIM + d] = (_Float16)val;
            } else {
                Vt[((size_t)bh * HDIM + d) * SEQ + s] = (_Float16)val;
            }
        }
    }
}

// ---------------------------------------------------------------------------
// Kernel 3: flash attention. One wave owns a 16-query tile of one (b,h) pair,
// streams 32-key chunks. Score WMMAs use K-fragments prefetched in the
// previous iteration; V-fragments are loaded immediately after the score
// WMMAs so they overlap the softmax VALU chain. P switches from C-fragment
// to A-fragment layout through a 1KB per-wave LDS staging tile.
// ---------------------------------------------------------------------------
__global__ void __launch_bounds__(128)
attn_kernel(const _Float16* __restrict__ Qh, const _Float16* __restrict__ Kh,
            const _Float16* __restrict__ Vt, const float* __restrict__ mask,
            float* __restrict__ out) {
    __shared__ _Float16 lds[4][16 * 32];            // 1 KB staging tile / wave

    int wib  = threadIdx.x >> 5;
    int gw   = blockIdx.x * 4 + wib;                // 0..4095
    int bh   = gw >> 6;                             // 0..63
    int qt   = gw & 63;                             // query tile 0..63
    int b    = bh >> 4, h = bh & 15;
    int lane = threadIdx.x & 31;
    int half = lane >> 4, r = lane & 15;
    _Float16* ptile = &lds[wib][0];

    // Q A-fragments for d=[0,32) and d=[32,64), loaded once (scale pre-folded)
    const _Float16* qrow = Qh + ((size_t)bh * SEQ + qt * 16 + r) * HDIM;
    v16h qa0 = cat16(ld8h(qrow + half * 16),      ld8h(qrow + half * 16 + 8));
    v16h qa1 = cat16(ld8h(qrow + 32 + half * 16), ld8h(qrow + 32 + half * 16 + 8));

    v8f O0 = zero8(), O1 = zero8(), O2 = zero8(), O3 = zero8();
    float mrun[8], lsum[8];
#pragma unroll
    for (int v = 0; v < 8; ++v) { mrun[v] = -1e30f; lsum[v] = 0.0f; }

    const _Float16* kbase = Kh + (size_t)bh * SEQ * HDIM;
    const _Float16* vbase = Vt + (size_t)bh * HDIM * SEQ;
    const float*    mrow  = mask + (size_t)b * SEQ;

    // --- pipeline prologue: K fragments for j = 0 ---
    const _Float16* kr0 = kbase + (size_t)(r) * HDIM;
    const _Float16* kr1 = kbase + (size_t)(16 + r) * HDIM;
    v16h kb00 = cat16(ld8h(kr0 + half * 16),      ld8h(kr0 + half * 16 + 8));
    v16h kb01 = cat16(ld8h(kr0 + 32 + half * 16), ld8h(kr0 + 32 + half * 16 + 8));
    v16h kb10 = cat16(ld8h(kr1 + half * 16),      ld8h(kr1 + half * 16 + 8));
    v16h kb11 = cat16(ld8h(kr1 + 32 + half * 16), ld8h(kr1 + 32 + half * 16 + 8));

    for (int j = 0; j < 32; ++j) {                  // 32 keys per iteration
        v8f sc0 = WMMA_F16(qa0, kb00, zero8());
        sc0     = WMMA_F16(qa1, kb01, sc0);
        v8f sc1 = WMMA_F16(qa0, kb10, zero8());
        sc1     = WMMA_F16(qa1, kb11, sc1);

        // V fragments for this iteration: independent of the softmax chain,
        // issue now so they overlap the exp/reduction VALU work.
        const _Float16* vp0 = vbase + (size_t)( 0 + r) * SEQ + j * 32 + half * 16;
        const _Float16* vp1 = vbase + (size_t)(16 + r) * SEQ + j * 32 + half * 16;
        const _Float16* vp2 = vbase + (size_t)(32 + r) * SEQ + j * 32 + half * 16;
        const _Float16* vp3 = vbase + (size_t)(48 + r) * SEQ + j * 32 + half * 16;
        v16h vb0 = cat16(ld8h(vp0), ld8h(vp0 + 8));
        v16h vb1 = cat16(ld8h(vp1), ld8h(vp1 + 8));
        v16h vb2 = cat16(ld8h(vp2), ld8h(vp2 + 8));
        v16h vb3 = cat16(ld8h(vp3), ld8h(vp3 + 8));

        // K fragments for next iteration ((j+1)&31 wraps harmlessly on last)
        {
            int jn = (j + 1) & 31;
            const _Float16* nr0 = kbase + (size_t)(jn * 32 + r) * HDIM;
            const _Float16* nr1 = kbase + (size_t)(jn * 32 + 16 + r) * HDIM;
            kb00 = cat16(ld8h(nr0 + half * 16),      ld8h(nr0 + half * 16 + 8));
            kb01 = cat16(ld8h(nr0 + 32 + half * 16), ld8h(nr0 + 32 + half * 16 + 8));
            kb10 = cat16(ld8h(nr1 + half * 16),      ld8h(nr1 + half * 16 + 8));
            kb11 = cat16(ld8h(nr1 + 32 + half * 16), ld8h(nr1 + 32 + half * 16 + 8));
        }

        float am0 = mrow[j * 32 + r];
        float am1 = mrow[j * 32 + 16 + r];

        float p0[8], p1[8];
#pragma unroll
        for (int v = 0; v < 8; ++v) {
            float s0 = sc0[v] + am0;
            float s1 = sc1[v] + am1;
            float t = fmaxf(s0, s1);
            // row max across the 16 lanes of this half-wave (C layout: lane=N)
            t = fmaxf(t, __shfl_xor(t, 1, 32));
            t = fmaxf(t, __shfl_xor(t, 2, 32));
            t = fmaxf(t, __shfl_xor(t, 4, 32));
            t = fmaxf(t, __shfl_xor(t, 8, 32));
            float mnew = fmaxf(mrun[v], t);
            p0[v] = __expf(s0 - mnew);
            p1[v] = __expf(s1 - mnew);
            float rs = p0[v] + p1[v];
            rs += __shfl_xor(rs, 1, 32);
            rs += __shfl_xor(rs, 2, 32);
            rs += __shfl_xor(rs, 4, 32);
            rs += __shfl_xor(rs, 8, 32);
            float alpha = __expf(mrun[v] - mnew);
            lsum[v] = lsum[v] * alpha + rs;
            mrun[v] = mnew;
            O0[v] *= alpha; O1[v] *= alpha; O2[v] *= alpha; O3[v] *= alpha;
        }

        // C layout -> LDS row-major [16 rows][32 keys]
#pragma unroll
        for (int v = 0; v < 8; ++v) {
            int row = v + half * 8;
            ptile[row * 32 + r]      = (_Float16)p0[v];
            ptile[row * 32 + 16 + r] = (_Float16)p1[v];
        }
        // wave-private tile: wait all DS stores, then reload in A-frag layout
        asm volatile("s_wait_dscnt 0" ::: "memory");
        v16h pa = cat16(*(const v8h*)&ptile[r * 32 + half * 16],
                        *(const v8h*)&ptile[r * 32 + half * 16 + 8]);
        asm volatile("" ::: "memory");   // pin next iteration's stores after loads

        // O += P @ V
        O0 = WMMA_F16(pa, vb0, O0);
        O1 = WMMA_F16(pa, vb1, O1);
        O2 = WMMA_F16(pa, vb2, O2);
        O3 = WMMA_F16(pa, vb3, O3);
    }

    // Normalize and store fp32 output [B, S, heads*HDIM]
#pragma unroll
    for (int v = 0; v < 8; ++v) {
        int row = v + half * 8;
        size_t obase = ((size_t)b * SEQ + qt * 16 + row) * HID + h * HDIM;
        float inv = 1.0f / lsum[v];
        out[obase + 0 * 16 + r] = O0[v] * inv;
        out[obase + 1 * 16 + r] = O1[v] * inv;
        out[obase + 2 * 16 + r] = O2[v] * inv;
        out[obase + 3 * 16 + r] = O3[v] * inv;
    }
}

// ---------------------------------------------------------------------------
// Launch
// ---------------------------------------------------------------------------
extern "C" void kernel_launch(void* const* d_in, const int* in_sizes, int n_in,
                              void* d_out, int out_size, void* d_ws, size_t ws_size,
                              hipStream_t stream) {
    const float* X    = (const float*)d_in[0];
    const float* mask = (const float*)d_in[1];
    const float* Wq   = (const float*)d_in[2];
    const float* bq   = (const float*)d_in[3];
    const float* Wk   = (const float*)d_in[4];
    const float* bk   = (const float*)d_in[5];
    const float* Wv   = (const float*)d_in[6];
    const float* bv   = (const float*)d_in[7];
    float* out = (float*)d_out;

    // Workspace layout (f16):
    //   Xh [4096][1024]          8 MB
    //   Wh [3][1024][1024]       6 MB
    //   Qh [64][1024][64]        8 MB
    //   Kh [64][1024][64]        8 MB
    //   Vt [64][64][1024]        8 MB   (V transposed per head)
    char* ws = (char*)d_ws;
    const size_t SZ_X = (size_t)MROWS * HID * 2;
    const size_t SZ_W = (size_t)3 * HID * HID * 2;
    const size_t SZ_Q = (size_t)BATCH * NHEAD * SEQ * HDIM * 2;
    _Float16* Xh = (_Float16*)(ws);
    _Float16* Wh = (_Float16*)(ws + SZ_X);
    _Float16* Qh = (_Float16*)(ws + SZ_X + SZ_W);
    _Float16* Kh = (_Float16*)(ws + SZ_X + SZ_W + SZ_Q);
    _Float16* Vt = (_Float16*)(ws + SZ_X + SZ_W + 2 * SZ_Q);

    cvt_kernel<<<dim3(1024), dim3(256), 0, stream>>>(X, Wq, Wk, Wv, Xh, Wh);

    // 256 M-tiles x 48 N-tiles(64-wide) = 12288 waves = 1536 blocks of 8 waves
    qkv_kernel<<<dim3(1536), dim3(256), 0, stream>>>(Xh, Wh, bq, bk, bv, Qh, Kh, Vt);

    // 64 (b,h) x 64 query-tiles = 4096 waves = 1024 blocks of 4 waves
    attn_kernel<<<dim3(1024), dim3(128), 0, stream>>>(Qh, Kh, Vt, mask, out);
}
